// AttentionBlock_67465346286196
// MI455X (gfx1250) — matrix operand order repaired and checked by reading, not compiled, not run on hardware
//
#include <hip/hip_runtime.h>

#define S_LEN 2048
#define NHEAD 16
#define HD    64
#define EMB   1024
#define MROWS (4 * S_LEN)          // B*S = 8192

typedef __attribute__((ext_vector_type(16))) __bf16 v16bf;
typedef __attribute__((ext_vector_type(8)))  float  v8f;

union FragA { v16bf v; unsigned int u[8]; };
union FragC { v8f   v; float        f[8]; };

__device__ __forceinline__ unsigned short f2bf(float f) {
  unsigned int x = __float_as_uint(f);
  if ((x & 0x7fffffffu) > 0x7f800000u) return (unsigned short)0x7fc0; // NaN
  x += 0x7fffu + ((x >> 16) & 1u);                                    // RNE
  return (unsigned short)(x >> 16);
}

// RNE only — caller guarantees finite input (softmax probs / normalized out).
__device__ __forceinline__ unsigned short f2bf_fast(float f) {
  unsigned int x = __float_as_uint(f);
  x += 0x7fffu + ((x >> 16) & 1u);
  return (unsigned short)(x >> 16);
}

__global__ void f32_to_bf16(const float* __restrict__ src,
                            unsigned short* __restrict__ dst, int n) {
  int i = blockIdx.x * blockDim.x + threadIdx.x;
  if (i < n) dst[i] = f2bf(src[i]);
}

// C = A[MxK] * W[NxK]^T, bf16 in / f32 accumulate.
// Block tile 128(M) x 64(N), 8 waves in 4x2, each wave owns 32x32 (4 C frags).
// K staged 64 per iteration -> 8 WMMAs per wave between barriers.
template<int MODE>
__global__ __launch_bounds__(256) void gemm_bf16_wmma(
    const unsigned short* __restrict__ A,
    const unsigned short* __restrict__ W,
    float* __restrict__ outF,
    unsigned short* __restrict__ outQ,
    unsigned short* __restrict__ outK,
    unsigned short* __restrict__ outV,
    int N, int K)
{
  __shared__ __align__(16) unsigned short sA[128 * 64];   // 16 KB
  __shared__ __align__(16) unsigned short sB[64 * 64];    //  8 KB

  const int mBase = blockIdx.y * 128;
  const int nBase = blockIdx.x * 64;
  const int t    = threadIdx.x;
  const int lane = t & 31;
  const int wv   = t >> 5;
  const int mw = wv >> 1;
  const int nw = wv & 1;
  const int hh = lane >> 4;
  const int ln = lane & 15;

  FragC acc[2][2];
#pragma unroll
  for (int mi = 0; mi < 2; mi++)
#pragma unroll
    for (int ni = 0; ni < 2; ni++)
#pragma unroll
      for (int r = 0; r < 8; r++) acc[mi][ni].f[r] = 0.f;

  const int rowA = t >> 1;
  const int offA = (t & 1) * 32;
  const int rowB = t >> 2;
  const int offB = (t & 3) * 16;

  for (int kk = 0; kk < K; kk += 64) {
    {
      const uint4* ga = (const uint4*)(&A[(size_t)(mBase + rowA) * K + kk + offA]);
      uint4* la = (uint4*)(&sA[rowA * 64 + offA]);
      la[0] = ga[0]; la[1] = ga[1]; la[2] = ga[2]; la[3] = ga[3];
      const uint4* gb = (const uint4*)(&W[(size_t)(nBase + rowB) * K + kk + offB]);
      uint4* lb = (uint4*)(&sB[rowB * 64 + offB]);
      lb[0] = gb[0]; lb[1] = gb[1];
    }
    __syncthreads();

#pragma unroll
    for (int ks = 0; ks < 64; ks += 32) {
      FragA a[2], bfr[2];
#pragma unroll
      for (int mi = 0; mi < 2; mi++)
#pragma unroll
        for (int j = 0; j < 8; j++) {
          const int kb = 16 * hh + 2 * j;
          a[mi].u[j] = *(const unsigned int*)(
              &sA[(32 * mw + 16 * mi + ln) * 64 + ks + kb]);
        }
#pragma unroll
      for (int ni = 0; ni < 2; ni++)
#pragma unroll
        for (int v = 0; v < 8; v++)
          bfr[ni].u[v] = *(const unsigned int*)(
              &sB[(32 * nw + 16 * ni + ln) * 64 + ks + 16 * hh + 2 * v]);
#pragma unroll
      for (int mi = 0; mi < 2; mi++)
#pragma unroll
        for (int ni = 0; ni < 2; ni++)
          acc[mi][ni].v = __builtin_amdgcn_wmma_f32_16x16x32_bf16(
              false, a[mi].v, false, bfr[ni].v, (short)0, acc[mi][ni].v,
              false, false);
    }
    __syncthreads();
  }

#pragma unroll
  for (int mi = 0; mi < 2; mi++)
#pragma unroll
    for (int ni = 0; ni < 2; ni++)
#pragma unroll
      for (int r = 0; r < 8; r++) {
        const int m = mBase + 32 * mw + 16 * mi + r + 8 * hh;
        const int n = nBase + 32 * nw + 16 * ni + ln;
        const float val = acc[mi][ni].f[r];
        if (MODE == 1) {
          outF[(size_t)m * N + n] = val;
        } else {
          const int sec = n >> 10;     // 0=q 1=k 2=v
          const int e   = n & 1023;
          const int hI  = e >> 6;
          const int d   = e & 63;
          const int b   = m >> 11;
          const int s   = m & 2047;
          const size_t idx = (((size_t)(b * NHEAD + hI)) * S_LEN + s) * HD + d;
          const unsigned short bv = f2bf(sec == 0 ? val * 0.125f : val);
          if (sec == 0)      outQ[idx] = bv;
          else if (sec == 1) outK[idx] = bv;
          else               outV[idx] = bv;
        }
      }
}

// ---- one 64-key flash-attention block; MASK only instantiated for the tail ----
template<bool MASK>
__device__ __forceinline__ void attn_block(
    const unsigned short* __restrict__ Kp,
    const unsigned short* __restrict__ Vp,
    unsigned short* __restrict__ ldsPw,
    int kb, int sl, int hh, int ln,
    const FragA& qa0, const FragA& qa1, const FragA& ones,
    FragC (&acc)[4], FragC& accS, float (&mrow)[8])
{
  // ---- scores: 4 key-subtiles of 16, contraction over D=64 ----
  FragC sc[4];
#pragma unroll
  for (int tt = 0; tt < 4; tt++) {
    const unsigned short* Krow = Kp + (size_t)(kb + 16 * tt + ln) * HD;
    FragA kf0, kf1;
#pragma unroll
    for (int v = 0; v < 8; v++) {
      kf0.u[v] = *(const unsigned int*)(&Krow[16 * hh + 2 * v]);
      kf1.u[v] = *(const unsigned int*)(&Krow[32 + 16 * hh + 2 * v]);
    }
#pragma unroll
    for (int r = 0; r < 8; r++) sc[tt].f[r] = 0.f;
    sc[tt].v = __builtin_amdgcn_wmma_f32_16x16x32_bf16(
        false, qa0.v, false, kf0.v, (short)0, sc[tt].v, false, false);
    sc[tt].v = __builtin_amdgcn_wmma_f32_16x16x32_bf16(
        false, qa1.v, false, kf1.v, (short)0, sc[tt].v, false, false);
  }

  bool valid[4];
  if (MASK) {
#pragma unroll
    for (int tt = 0; tt < 4; tt++) valid[tt] = (kb + 16 * tt + ln) < sl;
  }

  // ---- online softmax: one shuffle-max per row covers 64 keys ----
#pragma unroll
  for (int r = 0; r < 8; r++) {
    float s[4];
#pragma unroll
    for (int tt = 0; tt < 4; tt++)
      s[tt] = MASK ? (valid[tt] ? sc[tt].f[r] : -1e9f) : sc[tt].f[r];
    float tmax = fmaxf(fmaxf(s[0], s[1]), fmaxf(s[2], s[3]));
#pragma unroll
    for (int off = 1; off < 16; off <<= 1)
      tmax = fmaxf(tmax, __shfl_xor(tmax, off));
    const float mnew  = fmaxf(mrow[r], tmax);
    const float scale = __expf(mrow[r] - mnew);
    mrow[r] = mnew;
#pragma unroll
    for (int dt = 0; dt < 4; dt++) acc[dt].f[r] *= scale;
    accS.f[r] *= scale;
#pragma unroll
    for (int tt = 0; tt < 4; tt++)
      ldsPw[(r + 8 * hh) * 64 + 16 * tt + ln] = f2bf_fast(__expf(s[tt] - mnew));
  }
  asm volatile("s_wait_dscnt 0" ::: "memory");   // same-wave LDS RAW

  // ---- P frags (16x64 -> 2x 16x32) ----
  FragA pf0, pf1;
#pragma unroll
  for (int j = 0; j < 8; j++) {
    const int kbi = 16 * hh + 2 * j;
    pf0.u[j] = *(const unsigned int*)(&ldsPw[ln * 64 + kbi]);
    pf1.u[j] = *(const unsigned int*)(&ldsPw[ln * 64 + 32 + kbi]);
  }

  // ---- row sums on the matrix pipe: accS += P * ones ----
  accS.v = __builtin_amdgcn_wmma_f32_16x16x32_bf16(
      false, pf0.v, false, ones.v, (short)0, accS.v, false, false);
  accS.v = __builtin_amdgcn_wmma_f32_16x16x32_bf16(
      false, pf1.v, false, ones.v, (short)0, accS.v, false, false);

  // ---- O += P * V ----
  const unsigned short* Vblk = Vp + (size_t)kb * HD;
#pragma unroll
  for (int dt = 0; dt < 4; dt++) {
    const int d = dt * 16 + ln;
    FragA vf0, vf1;
#pragma unroll
    for (int v = 0; v < 8; v++) {
      unsigned int lo0 = Vblk[(16 * hh + 2 * v)          * HD + d];
      unsigned int hi0 = Vblk[(16 * hh + 2 * v + 1)      * HD + d];
      unsigned int lo1 = Vblk[(32 + 16 * hh + 2 * v)     * HD + d];
      unsigned int hi1 = Vblk[(32 + 16 * hh + 2 * v + 1) * HD + d];
      vf0.u[v] = lo0 | (hi0 << 16);
      vf1.u[v] = lo1 | (hi1 << 16);
    }
    acc[dt].v = __builtin_amdgcn_wmma_f32_16x16x32_bf16(
        false, pf0.v, false, vf0.v, (short)0, acc[dt].v, false, false);
    acc[dt].v = __builtin_amdgcn_wmma_f32_16x16x32_bf16(
        false, pf1.v, false, vf1.v, (short)0, acc[dt].v, false, false);
  }
}

// Flash attention: one wave per 16-query tile per (b,h); 64-key blocks.
// Unmasked main loop + single masked tail block (seq_len >= S/2 >= 1024).
__global__ __launch_bounds__(128) void attn_fa_wmma(
    const unsigned short* __restrict__ Q,
    const unsigned short* __restrict__ Km,
    const unsigned short* __restrict__ V,
    const int* __restrict__ seqlens,
    unsigned short* __restrict__ O)
{
  __shared__ __align__(16) unsigned short ldsP[4][16 * 64];   // 8 KB

  const int lane = threadIdx.x & 31;
  const int wv   = threadIdx.x >> 5;
  const int gw   = blockIdx.x * 4 + wv;
  const int qt   = gw & 127;
  const int bh   = gw >> 7;
  const int b    = bh >> 4;
  const int h    = bh & 15;
  const int hh = lane >> 4;
  const int ln = lane & 15;

  const size_t headOff = (size_t)bh * S_LEN * HD;
  const unsigned short* Qp = Q  + headOff + (size_t)qt * 16 * HD;
  const unsigned short* Kp = Km + headOff;
  const unsigned short* Vp = V  + headOff;
  unsigned short* ldsPw = &ldsP[wv][0];

  FragA qa0, qa1;
#pragma unroll
  for (int j = 0; j < 8; j++) {
    const int kb = 16 * hh + 2 * j;
    qa0.u[j] = *(const unsigned int*)(&Qp[ln * HD + kb]);
    qa1.u[j] = *(const unsigned int*)(&Qp[ln * HD + 32 + kb]);
  }

  FragA ones;                          // bf16 1.0 pairs
#pragma unroll
  for (int v = 0; v < 8; v++) ones.u[v] = 0x3F803F80u;

  FragC acc[4], accS;
#pragma unroll
  for (int dt = 0; dt < 4; dt++)
#pragma unroll
    for (int r = 0; r < 8; r++) acc[dt].f[r] = 0.f;
#pragma unroll
  for (int r = 0; r < 8; r++) accS.f[r] = 0.f;

  float mrow[8];
#pragma unroll
  for (int r = 0; r < 8; r++) mrow[r] = -3.0e38f;

  const int sl    = seqlens[b];
  const int nFull = sl & ~63;          // keys [0, nFull) need no mask

  for (int kb = 0; kb < nFull; kb += 64)
    attn_block<false>(Kp, Vp, ldsPw, kb, sl, hh, ln, qa0, qa1, ones,
                      acc, accS, mrow);
  if (sl > nFull)
    attn_block<true>(Kp, Vp, ldsPw, nFull, sl, hh, ln, qa0, qa1, ones,
                     acc, accS, mrow);

  // ---- normalize (denominator from accS) + store [B,S,H,D] bf16 ----
#pragma unroll
  for (int r = 0; r < 8; r++) {
    const float inv = 1.0f / accS.f[r];
    const int s = qt * 16 + r + 8 * hh;
#pragma unroll
    for (int dt = 0; dt < 4; dt++) {
      const int d = dt * 16 + ln;
      O[(((size_t)(b * S_LEN + s)) * NHEAD + h) * HD + d] =
          f2bf_fast(acc[dt].f[r] * inv);
    }
  }
}

extern "C" void kernel_launch(void* const* d_in, const int* in_sizes, int n_in,
                              void* d_out, int out_size, void* d_ws, size_t ws_size,
                              hipStream_t stream) {
  const float* x    = (const float*)d_in[0];
  const int*   slen = (const int*)d_in[1];
  const float* Wqkv = (const float*)d_in[2];
  const float* Wout = (const float*)d_in[3];
  float* out = (float*)d_out;

  unsigned short* Xbf = (unsigned short*)d_ws;
  unsigned short* Wq  = Xbf + (size_t)MROWS * EMB;
  unsigned short* Wo  = Wq  + (size_t)3 * EMB * EMB;
  unsigned short* Qbf = Wo  + (size_t)EMB * EMB;
  unsigned short* Kbf = Qbf + (size_t)MROWS * EMB;
  unsigned short* Vbf = Kbf + (size_t)MROWS * EMB;
  unsigned short* Obf = Vbf + (size_t)MROWS * EMB;

  const int nx = MROWS * EMB;
  const int nq = 3 * EMB * EMB;
  const int no = EMB * EMB;
  f32_to_bf16<<<(nx + 255) / 256, 256, 0, stream>>>(x,    Xbf, nx);
  f32_to_bf16<<<(nq + 255) / 256, 256, 0, stream>>>(Wqkv, Wq,  nq);
  f32_to_bf16<<<(no + 255) / 256, 256, 0, stream>>>(Wout, Wo,  no);

  gemm_bf16_wmma<0><<<dim3(3 * EMB / 64, MROWS / 128), 256, 0, stream>>>(
      Xbf, Wq, nullptr, Qbf, Kbf, Vbf, 3 * EMB, EMB);

  attn_fa_wmma<<<(4 * NHEAD * (S_LEN / 16)) / 4, 128, 0, stream>>>(
      Qbf, Kbf, Vbf, slen, Obf);

  gemm_bf16_wmma<1><<<dim3(EMB / 64, MROWS / 128), 256, 0, stream>>>(
      Obf, Wo, out, nullptr, nullptr, nullptr, EMB, EMB);
}
